// CrossAttention_25211458027991
// MI455X (gfx1250) — matrix operand order repaired and checked by reading, not compiled, not run on hardware
//
#include <hip/hip_runtime.h>

typedef __attribute__((ext_vector_type(16))) _Float16 v16h;
typedef __attribute__((ext_vector_type(8)))  _Float16 v8h;
typedef __attribute__((ext_vector_type(8)))  float    v8f;

#define NTOK 4096
#define CDIM 256
#define KCONV 2304   // 256*9
#define SCALE 0.17677669529663687f  // 32^-0.5

// ---------------------------------------------------------------------------
// helpers
// ---------------------------------------------------------------------------

// A-fragment (16x32 f16, row-major source with row already selected):
// lane l=L&15 holds row l; e0..15 = K[hi*16 .. +15]
__device__ inline v16h load_a_frag(const _Float16* p, int k0, int hi) {
    v8h lo = *(const v8h*)(p + k0 + hi * 16);
    v8h hv = *(const v8h*)(p + k0 + hi * 16 + 8);
    return __builtin_shufflevector(lo, hv, 0,1,2,3,4,5,6,7,8,9,10,11,12,13,14,15);
}

__device__ inline v8f wmma_f16(v16h a, v16h b, v8f c) {
    return __builtin_amdgcn_wmma_f32_16x16x32_f16(false, a, false, b, (short)0, c,
                                                  false, false);
}

// Async copy of 16 bytes global -> LDS (CDNA5 async-LDS path, ASYNCcnt).
// vdst operand supplies the per-lane LDS byte address.
__device__ inline void async_ld16(void* lds_ptr, const _Float16* g) {
    unsigned loff =
        (unsigned)(unsigned long long)(__attribute__((address_space(3))) void*)lds_ptr;
    asm volatile("global_load_async_to_lds_b128 %0, %1, off"
                 :: "v"(loff), "v"((unsigned long long)g)
                 : "memory");
}

__device__ inline void wait_async0() {
    asm volatile("s_wait_asynccnt 0" ::: "memory");
}

// ---------------------------------------------------------------------------
// fp32 -> f16 convert
// ---------------------------------------------------------------------------
__global__ void cvt_f16_kernel(const float* __restrict__ src,
                               _Float16* __restrict__ dst, int n) {
    int i = blockIdx.x * 256 + threadIdx.x;
    if (i < n) dst[i] = (_Float16)src[i];
}

// ---------------------------------------------------------------------------
// im2col for 3x3 SAME conv on [256][64][64] fp32 -> [4096][2304] f16
// col index k = ic*9 + ky*3 + kx
// ---------------------------------------------------------------------------
__global__ void im2col_kernel(const float* __restrict__ in,
                              _Float16* __restrict__ out) {
    int idx = blockIdx.x * 256 + threadIdx.x;       // < 4096*2304
    int t = idx / KCONV;
    int k = idx - t * KCONV;
    int ic = k / 9;
    int r  = k - ic * 9;
    int ky = r / 3, kx = r - ky * 3;
    int y = t >> 6, x = t & 63;
    int yy = y + ky - 1, xx = x + kx - 1;
    float v = 0.0f;
    if ((unsigned)yy < 64u && (unsigned)xx < 64u)
        v = in[(ic << 12) + (yy << 6) + xx];
    out[idx] = (_Float16)v;
}

// ---------------------------------------------------------------------------
// Generic TN GEMM:  D[M,N] = A[M,K] (row-major) * B[N,K]^T (row-major NxK) + bias
// 256 threads = 8 waves; block tile 128x64; wave tile 32x32 (2x2 WMMA).
// A/B tiles double-buffered in LDS, filled with global_load_async_to_lds_b128.
// Output modes:
//   0: f32 row-major   out[row*N + col]
//   1: f16 row-major   out[row*N + col]
//   2: q/k padded store: f16 out[(col>>4)*M*32 + row*32 + (col&15)], zero at +16,
//      value scaled by `scale` after bias
//   3: f16 col-major   out[col*M + row]   (v transposed per head)
//   4: f32 col-major   out[col*M + row]   (final [C][N] output)
// ---------------------------------------------------------------------------
__device__ inline void store_elem(void* out, int mode, int M, int N,
                                  int row, int col, float v) {
    switch (mode) {
        case 0: ((float*)out)[(size_t)row * N + col] = v; break;
        case 1: ((_Float16*)out)[(size_t)row * N + col] = (_Float16)v; break;
        case 2: {
            int h = col >> 4, d = col & 15;
            _Float16* o = (_Float16*)out + ((size_t)h * M + row) * 32 + d;
            o[0]  = (_Float16)v;
            o[16] = (_Float16)0.0f;
            break;
        }
        case 3: ((_Float16*)out)[(size_t)col * M + row] = (_Float16)v; break;
        case 4: ((float*)out)[(size_t)col * M + row] = v; break;
    }
}

__global__ void __launch_bounds__(256)
gemm_tn_wmma(const _Float16* __restrict__ A, int lda,
             const _Float16* __restrict__ B, int ldb,
             const float* __restrict__ bias,
             int M, int N, int K,
             void* __restrict__ out, int mode, float scale) {
    // LDS tiles: A 128x32, B 64x32, double-buffered (24 KB total)
    __shared__ __align__(32) _Float16 ldsA[2][128 * 32];
    __shared__ __align__(32) _Float16 ldsB[2][64 * 32];

    const int tid  = threadIdx.x;
    const int lane = tid & 31;
    const int wave = tid >> 5;
    const int wm = wave & 3, wn = wave >> 2;
    const int mBlk = blockIdx.x * 128;
    const int nBlk = blockIdx.y * 64;
    const int l = lane & 15, hi = lane >> 4;

    // staging assignment (16B chunks): A has 512 chunks, B has 256 chunks
    const int ar0 = tid >> 2,          as0 = (tid & 3) * 8;
    const int ar1 = (tid + 256) >> 2,  as1 = as0;      // same seg pattern
    const int bc  = tid >> 2,          bs  = (tid & 3) * 8;
    const _Float16* agp0 = A + (size_t)(mBlk + ar0) * lda + as0;
    const _Float16* agp1 = A + (size_t)(mBlk + ar1) * lda + as1;
    const _Float16* bgp  = B + (size_t)(nBlk + bc)  * ldb + bs;

    v8f c00 = {}, c01 = {}, c10 = {}, c11 = {};

    // prologue: stage k=0 into buffer 0
    async_ld16(&ldsA[0][ar0 * 32 + as0], agp0);
    async_ld16(&ldsA[0][ar1 * 32 + as1], agp1);
    async_ld16(&ldsB[0][bc  * 32 + bs],  bgp);

    int buf = 0;
    for (int k = 0; k < K; k += 32) {
        wait_async0();
        __syncthreads();

        if (k + 32 < K) {
            int nb = buf ^ 1;
            async_ld16(&ldsA[nb][ar0 * 32 + as0], agp0 + k + 32);
            async_ld16(&ldsA[nb][ar1 * 32 + as1], agp1 + k + 32);
            async_ld16(&ldsB[nb][bc  * 32 + bs],  bgp  + k + 32);
        }

        const _Float16* la = ldsA[buf];
        const _Float16* lb = ldsB[buf];
        v16h a0 = load_a_frag(la + (wm * 32 + l) * 32, 0, hi);
        v16h a1 = load_a_frag(la + (wm * 32 + 16 + l) * 32, 0, hi);
        v16h b0 = *(const v16h*)(lb + (wn * 32 + l) * 32 + hi * 16);
        v16h b1 = *(const v16h*)(lb + (wn * 32 + 16 + l) * 32 + hi * 16);
        c00 = wmma_f16(a0, b0, c00);
        c01 = wmma_f16(a0, b1, c01);
        c10 = wmma_f16(a1, b0, c10);
        c11 = wmma_f16(a1, b1, c11);

        __syncthreads();   // reads of buf done before it is restaged next iter
        buf ^= 1;
    }

    int colA = nBlk + wn * 32 + l;
    int colB = colA + 16;
    int mW   = mBlk + wm * 32;
    float bA = bias ? bias[colA] : 0.0f;
    float bB = bias ? bias[colB] : 0.0f;
#pragma unroll
    for (int r = 0; r < 8; ++r) {
        int rowA = mW + r + 8 * hi;
        int rowB = mW + 16 + r + 8 * hi;
        store_elem(out, mode, M, N, rowA, colA, (c00[r] + bA) * scale);
        store_elem(out, mode, M, N, rowA, colB, (c01[r] + bB) * scale);
        store_elem(out, mode, M, N, rowB, colA, (c10[r] + bA) * scale);
        store_elem(out, mode, M, N, rowB, colB, (c11[r] + bB) * scale);
    }
}

// ---------------------------------------------------------------------------
// Flash attention, 8 heads, qk-dim padded to 32 (upper 16 zero), v-dim 32.
//   qp: [8][4096][32] f16 (already scaled)      kp: [8][4096][32] f16
//   vt: [8][32][4096] f16 (v transposed)        o : [8][4096][32] f16
// block = 256 threads = 8 waves; each wave owns 16 queries; grid (32, 8).
// ---------------------------------------------------------------------------
__global__ void __launch_bounds__(256)
flash_attn_wmma(const _Float16* __restrict__ qp,
                const _Float16* __restrict__ kp,
                const _Float16* __restrict__ vt,
                _Float16* __restrict__ o) {
    __shared__ __align__(32) _Float16 slab[8 * 16 * 32];  // per-wave P staging

    int lane = threadIdx.x & 31;
    int wave = threadIdx.x >> 5;
    int l = lane & 15, hi = lane >> 4;
    int h  = blockIdx.y;
    int q0 = blockIdx.x * 128 + wave * 16;
    int sbase = wave * 512;

    const _Float16* kb = kp + (size_t)h * NTOK * 32;
    const _Float16* vb = vt + (size_t)h * 32 * NTOK;

    // Q fragment (16 queries x 32 padded dims), loaded once
    v16h qf = load_a_frag(qp + ((size_t)h * NTOK + q0 + l) * 32, 0, hi);

    v8f acc0 = {}, acc1 = {};
    float mrow[8], lrow[8];
#pragma unroll
    for (int r = 0; r < 8; ++r) { mrow[r] = -1e30f; lrow[r] = 0.0f; }

    for (int j = 0; j < NTOK; j += 32) {
        v16h kf0 = *(const v16h*)(kb + (size_t)(j + l)      * 32 + hi * 16);
        v16h kf1 = *(const v16h*)(kb + (size_t)(j + 16 + l) * 32 + hi * 16);
        v8f z = {};
        v8f s0 = wmma_f16(qf, kf0, z);   // 16 queries x keys j..j+15
        v8f s1 = wmma_f16(qf, kf1, z);   // keys j+16..j+31

#pragma unroll
        for (int r = 0; r < 8; ++r) {
            // row max across the 16 lanes of this half-wave
            float t = fmaxf(s0[r], s1[r]);
            t = fmaxf(t, __shfl_xor(t, 1, 32));
            t = fmaxf(t, __shfl_xor(t, 2, 32));
            t = fmaxf(t, __shfl_xor(t, 4, 32));
            t = fmaxf(t, __shfl_xor(t, 8, 32));
            float mn   = fmaxf(mrow[r], t);
            float corr = __expf(mrow[r] - mn);
            float p0 = __expf(s0[r] - mn);
            float p1 = __expf(s1[r] - mn);
            float sm = p0 + p1;
            sm += __shfl_xor(sm, 1, 32);
            sm += __shfl_xor(sm, 2, 32);
            sm += __shfl_xor(sm, 4, 32);
            sm += __shfl_xor(sm, 8, 32);
            lrow[r] = lrow[r] * corr + sm;
            mrow[r] = mn;
            acc0[r] *= corr;
            acc1[r] *= corr;
            // stage P (C-layout row r+8*hi, col l) into LDS in [row][key] order
            slab[sbase + (r + 8 * hi) * 32 + l]      = (_Float16)p0;
            slab[sbase + (r + 8 * hi) * 32 + 16 + l] = (_Float16)p1;
        }
        asm volatile("s_wait_dscnt 0" ::: "memory");

        // reload P as an A-fragment (16 queries x 32 keys)
        v16h pf = load_a_frag(&slab[sbase + l * 32], 0, hi);

        // V tiles: B layout, lane c = v-dim column, contiguous keys
        v16h vf0 = *(const v16h*)(vb + (size_t)l        * NTOK + j + hi * 16);
        v16h vf1 = *(const v16h*)(vb + (size_t)(16 + l) * NTOK + j + hi * 16);
        acc0 = wmma_f16(pf, vf0, acc0);
        acc1 = wmma_f16(pf, vf1, acc1);
    }

    _Float16* ob = o + (size_t)h * NTOK * 32;
#pragma unroll
    for (int r = 0; r < 8; ++r) {
        int row = q0 + r + 8 * hi;
        float inv = 1.0f / lrow[r];
        ob[(size_t)row * 32 + l]      = (_Float16)(acc0[r] * inv);
        ob[(size_t)row * 32 + 16 + l] = (_Float16)(acc1[r] * inv);
    }
}

// ---------------------------------------------------------------------------
// host launch
// ---------------------------------------------------------------------------
extern "C" void kernel_launch(void* const* d_in, const int* in_sizes, int n_in,
                              void* d_out, int out_size, void* d_ws, size_t ws_size,
                              hipStream_t stream) {
    (void)in_sizes; (void)n_in; (void)out_size; (void)ws_size;

    const float* x   = (const float*)d_in[0];
    const float* akv = (const float*)d_in[1];
    const float* qw  = (const float*)d_in[2];
    const float* qb  = (const float*)d_in[3];
    const float* kvw = (const float*)d_in[4];
    const float* kvb = (const float*)d_in[5];
    const float* Wq  = (const float*)d_in[6];
    const float* bq  = (const float*)d_in[7];
    const float* Wk  = (const float*)d_in[8];
    const float* bk  = (const float*)d_in[9];
    const float* Wv  = (const float*)d_in[10];
    const float* bv  = (const float*)d_in[11];
    const float* Wp  = (const float*)d_in[12];
    const float* bp  = (const float*)d_in[13];

    char* ws = (char*)d_ws;
    auto alloc = [&](size_t bytes) -> char* {
        char* p = ws;
        ws += (bytes + 255) & ~(size_t)255;
        return p;
    };
    _Float16* colbuf = (_Float16*)alloc((size_t)NTOK * KCONV * 2);  // 18.9 MB
    _Float16* wq_h   = (_Float16*)alloc((size_t)CDIM * KCONV * 2);
    _Float16* wkv_h  = (_Float16*)alloc((size_t)CDIM * KCONV * 2);
    _Float16* Wq_h   = (_Float16*)alloc((size_t)128 * CDIM * 2);
    _Float16* Wk_h   = (_Float16*)alloc((size_t)128 * CDIM * 2);
    _Float16* Wv_h   = (_Float16*)alloc((size_t)CDIM * CDIM * 2);
    _Float16* Wp_h   = (_Float16*)alloc((size_t)CDIM * CDIM * 2);
    _Float16* xf     = (_Float16*)alloc((size_t)NTOK * CDIM * 2);
    _Float16* kvf    = (_Float16*)alloc((size_t)NTOK * CDIM * 2);
    _Float16* qpad   = (_Float16*)alloc((size_t)8 * NTOK * 32 * 2);
    _Float16* kpad   = (_Float16*)alloc((size_t)8 * NTOK * 32 * 2);
    _Float16* vtr    = (_Float16*)alloc((size_t)8 * 32 * NTOK * 2);
    _Float16* obuf   = (_Float16*)alloc((size_t)8 * NTOK * 32 * 2);

    auto cvt = [&](const float* s, _Float16* d, int n) {
        cvt_f16_kernel<<<(n + 255) / 256, 256, 0, stream>>>(s, d, n);
    };
    cvt(qw,  wq_h,  CDIM * KCONV);
    cvt(kvw, wkv_h, CDIM * KCONV);
    cvt(Wq,  Wq_h,  128 * CDIM);
    cvt(Wk,  Wk_h,  128 * CDIM);
    cvt(Wv,  Wv_h,  CDIM * CDIM);
    cvt(Wp,  Wp_h,  CDIM * CDIM);

    const int IM2COL_BLOCKS = (NTOK * KCONV) / 256;  // 36864

    // conv q: im2col(x) -> GEMM with qw  => xf [4096x256] f16
    im2col_kernel<<<IM2COL_BLOCKS, 256, 0, stream>>>(x, colbuf);
    gemm_tn_wmma<<<dim3(NTOK / 128, CDIM / 64), 256, 0, stream>>>(
        colbuf, KCONV, wq_h, KCONV, qb, NTOK, CDIM, KCONV, xf, 1, 1.0f);

    // conv kv: im2col(attn_kv) -> GEMM with kvw => kvf [4096x256] f16
    im2col_kernel<<<IM2COL_BLOCKS, 256, 0, stream>>>(akv, colbuf);
    gemm_tn_wmma<<<dim3(NTOK / 128, CDIM / 64), 256, 0, stream>>>(
        colbuf, KCONV, wkv_h, KCONV, kvb, NTOK, CDIM, KCONV, kvf, 1, 1.0f);

    // q = (xf @ Wq^T + bq) * SCALE -> padded [8][4096][32]
    gemm_tn_wmma<<<dim3(NTOK / 128, 128 / 64), 256, 0, stream>>>(
        xf, CDIM, Wq_h, CDIM, bq, NTOK, 128, CDIM, qpad, 2, SCALE);
    // k = kvf @ Wk^T + bk -> padded [8][4096][32]
    gemm_tn_wmma<<<dim3(NTOK / 128, 128 / 64), 256, 0, stream>>>(
        kvf, CDIM, Wk_h, CDIM, bk, NTOK, 128, CDIM, kpad, 2, 1.0f);
    // v = kvf @ Wv^T + bv -> transposed [8][32][4096]
    gemm_tn_wmma<<<dim3(NTOK / 128, CDIM / 64), 256, 0, stream>>>(
        kvf, CDIM, Wv_h, CDIM, bv, NTOK, CDIM, CDIM, vtr, 3, 1.0f);

    // attention -> obuf [8][4096][32] f16 (== scrambled [4096][256] A matrix)
    flash_attn_wmma<<<dim3(NTOK / 128, 8), 256, 0, stream>>>(qpad, kpad, vtr, obuf);

    // out[c][t] = (obuf @ Wp^T + bp)[t][c]  -> d_out as [256][64][64] fp32
    gemm_tn_wmma<<<dim3(NTOK / 128, CDIM / 64), 256, 0, stream>>>(
        obuf, CDIM, Wp_h, CDIM, bp, NTOK, CDIM, CDIM, d_out, 4, 1.0f);
}
